// GemNet_49984829391024
// MI455X (gfx1250) — matrix operand order repaired and checked by reading, not compiled, not run on hardware
//
#include <hip/hip_runtime.h>
#include <hip/hip_bf16.h>

typedef __bf16 bf16_t;
typedef __attribute__((ext_vector_type(16))) __bf16 v16bf;
typedef __attribute__((ext_vector_type(8)))  __bf16 v8bf;
typedef __attribute__((ext_vector_type(8)))  float  v8f;

#define NNODE  50000
#define NEDGE  800000
#define NQUAD  2000000
#define HDIM   128
#define NGRAPH 64

__device__ inline v8f vzero8() {
  v8f z;
#pragma unroll
  for (int i = 0; i < 8; ++i) z[i] = 0.f;
  return z;
}

// A fragment: 16x32 bf16 tile row-major in LDS (stride ldt elements).
// lanes 0-15: row=lane, K = kk*32 + {0..7, 16..23}
// lanes 16-31: row=lane-16, K = kk*32 + {8..15, 24..31}
__device__ inline v16bf load_a_frag(const bf16_t* tile, int ldt, int kk) {
  const int lane  = threadIdx.x & 31;
  const int row   = lane & 15;
  const int khalf = lane >> 4;
  const bf16_t* p = tile + row * ldt + kk * 32 + khalf * 8;
  v8bf lo = *(const v8bf*)(p);
  v8bf hi = *(const v8bf*)(p + 16);
  return __builtin_shufflevector(lo, hi, 0,1,2,3,4,5,6,7,8,9,10,11,12,13,14,15);
}

// B fragment from pre-packed weights: contiguous 16 bf16 per lane.
__device__ inline v16bf load_b_frag(const bf16_t* Wp, int kk, int t) {
  const int lane = threadIdx.x & 31;
  return *(const v16bf*)(Wp + ((((kk * 8) + t) * 32 + lane) << 4));
}

// Fused (Linear -> ReLU -> Linear) on a 16-row tile living in LDS.
// Input: tile[16][KCH1*32] bf16. Hidden (bf16, ReLU'd, +b1) is written back
// into tile cols 0..127. Output (f32, +b2) left in D-fragment form in out[8].
// B fragments for each K-chunk are batch-loaded into independent registers so
// the 16 global_load_b128 can fly in one clause instead of serializing on
// s_wait_loadcnt 0 before each WMMA.
template<int KCH1>
__device__ inline void mlp2_tile(bf16_t* tile, int ldt,
                                 const bf16_t* W1p, const float* b1,
                                 const bf16_t* W2p, const float* b2,
                                 v8f out[8]) {
  const int lane = threadIdx.x & 31;
  v8f acc[8];
#pragma unroll
  for (int t = 0; t < 8; ++t) acc[t] = vzero8();
#pragma unroll
  for (int kk = 0; kk < KCH1; ++kk) {
    v16bf a = load_a_frag(tile, ldt, kk);
    v16bf bfr[8];
#pragma unroll
    for (int t = 0; t < 8; ++t) bfr[t] = load_b_frag(W1p, kk, t);
#pragma unroll
    for (int t = 0; t < 8; ++t)
      acc[t] = __builtin_amdgcn_wmma_f32_16x16x32_bf16(false, a, false, bfr[t],
                                                       (short)0, acc[t], false, false);
  }
  // bias + ReLU, hidden back to LDS as bf16 (same-wave LDS ops are in-order)
#pragma unroll
  for (int t = 0; t < 8; ++t) {
    const int col  = t * 16 + (lane & 15);
    const float bs = b1[col];
#pragma unroll
    for (int e = 0; e < 8; ++e) {
      const int row = (lane < 16) ? e : (8 + e);
      float v = acc[t][e] + bs;
      v = v > 0.f ? v : 0.f;
      tile[row * ldt + col] = (bf16_t)v;
    }
  }
  __builtin_amdgcn_wave_barrier();
#pragma unroll
  for (int t = 0; t < 8; ++t) out[t] = vzero8();
#pragma unroll
  for (int kk = 0; kk < 4; ++kk) {
    v16bf a = load_a_frag(tile, ldt, kk);
    v16bf bfr[8];
#pragma unroll
    for (int t = 0; t < 8; ++t) bfr[t] = load_b_frag(W2p, kk, t);
#pragma unroll
    for (int t = 0; t < 8; ++t)
      out[t] = __builtin_amdgcn_wmma_f32_16x16x32_bf16(false, a, false, bfr[t],
                                                       (short)0, out[t], false, false);
  }
#pragma unroll
  for (int t = 0; t < 8; ++t) {
    const int col  = t * 16 + (lane & 15);
    const float bs = b2[col];
#pragma unroll
    for (int e = 0; e < 8; ++e) out[t][e] += bs;
  }
}

// Pack f32 W[K x 128] into WMMA B-fragment order, zero-padded to KCH*32 rows.
__global__ void pack_w_kernel(const float* __restrict__ W, bf16_t* __restrict__ out,
                              int K, int KCH) {
  int idx = blockIdx.x * 256 + threadIdx.x;
  int total = KCH << 12;            // KCH*8*32*16
  if (idx >= total) return;
  int i    = idx & 15;
  int lane = (idx >> 4) & 31;
  int t    = (idx >> 9) & 7;
  int kk   = idx >> 12;
  int col  = t * 16 + (lane & 15);
  int k    = kk * 32 + ((lane >> 4) * 8) + (i & 7) + ((i >> 3) * 16);
  float v  = (k < K) ? W[(size_t)k * HDIM + col] : 0.f;
  out[idx] = (bf16_t)v;
}

// ---- quadruplet message kernel: gather -> MLP2 -> atomic scatter (by k_idx)
__global__ __launch_bounds__(128) void quad_kernel(
    const float* __restrict__ h, const float* __restrict__ rbf,
    const float* __restrict__ cbf, const float* __restrict__ sbf,
    const int* __restrict__ l_idx, const int* __restrict__ k_idx,
    const bf16_t* __restrict__ W1p, const float* __restrict__ b1,
    const bf16_t* __restrict__ W2p, const float* __restrict__ b2,
    float* __restrict__ aggE, int Q) {
  __shared__ __align__(32) bf16_t smem[4][16 * 160];
  __shared__ int kidx_s[4][16];
  const int wave = threadIdx.x >> 5;
  const int lane = threadIdx.x & 31;
  const int qbase = (blockIdx.x * 4 + wave) * 16;
  if (qbase >= Q) return;                       // wave-uniform
  bf16_t* tile = smem[wave];
#pragma unroll 4
  for (int r = 0; r < 16; ++r) {
    const int q  = qbase + r;
    const int li = l_idx[q];
    const float4 hv = *(const float4*)(h + (size_t)li * HDIM + lane * 4);
    bf16_t* dp = tile + r * 160 + lane * 4;
    dp[0] = (bf16_t)hv.x; dp[1] = (bf16_t)hv.y;
    dp[2] = (bf16_t)hv.z; dp[3] = (bf16_t)hv.w;
  }
  if (lane < 16) {
    const int q  = qbase + lane;
    const int ki = k_idx[q];
    kidx_s[wave][lane] = ki;
    bf16_t* row = tile + lane * 160;
#pragma unroll
    for (int j = 0; j < 6; ++j) row[128 + j] = (bf16_t)rbf[(size_t)ki * 6 + j];
#pragma unroll
    for (int j = 0; j < 6; ++j) row[134 + j] = (bf16_t)cbf[(size_t)q * 6 + j];
#pragma unroll
    for (int j = 0; j < 6; ++j) row[140 + j] = (bf16_t)sbf[(size_t)q * 6 + j];
#pragma unroll
    for (int j = 146; j < 160; ++j) row[j] = (bf16_t)0.f;
  }
  v8f out[8];
  mlp2_tile<5>(tile, 160, W1p, b1, W2p, b2, out);
  // pull the 8 scatter rows' indices into registers once
  int kis[8];
#pragma unroll
  for (int e = 0; e < 8; ++e) kis[e] = kidx_s[wave][(lane < 16) ? e : (8 + e)];
#pragma unroll
  for (int t = 0; t < 8; ++t) {
    const int col = t * 16 + (lane & 15);
#pragma unroll
    for (int e = 0; e < 8; ++e) {
      __hip_atomic_fetch_add(&aggE[(size_t)kis[e] * HDIM + col], out[t][e],
                             __ATOMIC_RELAXED, __HIP_MEMORY_SCOPE_AGENT);
    }
  }
}

// ---- edge kernel: concat(h[src], aggE[dst]) -> MLP2 -> atomic scatter by dst
__global__ __launch_bounds__(128) void edge_kernel(
    const float* __restrict__ h, const float* __restrict__ aggE,
    const int* __restrict__ src, const int* __restrict__ dst,
    const bf16_t* __restrict__ W1p, const float* __restrict__ b1,
    const bf16_t* __restrict__ W2p, const float* __restrict__ b2,
    float* __restrict__ aggN, int E) {
  __shared__ __align__(32) bf16_t smem[4][16 * 256];
  __shared__ int dst_s[4][16];
  const int wave = threadIdx.x >> 5;
  const int lane = threadIdx.x & 31;
  const int ebase = (blockIdx.x * 4 + wave) * 16;
  if (ebase >= E) return;
  bf16_t* tile = smem[wave];
#pragma unroll 4
  for (int r = 0; r < 16; ++r) {
    const int e = ebase + r;
    const int s = src[e];
    const int d = dst[e];
    const float4 hv = *(const float4*)(h    + (size_t)s * HDIM + lane * 4);
    const float4 av = *(const float4*)(aggE + (size_t)d * HDIM + lane * 4);
    bf16_t* dp = tile + r * 256 + lane * 4;
    dp[0] = (bf16_t)hv.x; dp[1] = (bf16_t)hv.y;
    dp[2] = (bf16_t)hv.z; dp[3] = (bf16_t)hv.w;
    dp[128] = (bf16_t)av.x; dp[129] = (bf16_t)av.y;
    dp[130] = (bf16_t)av.z; dp[131] = (bf16_t)av.w;
  }
  if (lane < 16) dst_s[wave][lane] = dst[ebase + lane];
  v8f out[8];
  mlp2_tile<8>(tile, 256, W1p, b1, W2p, b2, out);
  int dis[8];
#pragma unroll
  for (int e = 0; e < 8; ++e) dis[e] = dst_s[wave][(lane < 16) ? e : (8 + e)];
#pragma unroll
  for (int t = 0; t < 8; ++t) {
    const int col = t * 16 + (lane & 15);
#pragma unroll
    for (int e = 0; e < 8; ++e) {
      __hip_atomic_fetch_add(&aggN[(size_t)dis[e] * HDIM + col], out[t][e],
                             __ATOMIC_RELAXED, __HIP_MEMORY_SCOPE_AGENT);
    }
  }
}

// ---- node update: concat(h, aggr) -> MLP2 -> h_next (dense store)
__global__ __launch_bounds__(128) void node_kernel(
    const float* __restrict__ h, const float* __restrict__ aggN,
    const bf16_t* __restrict__ W1p, const float* __restrict__ b1,
    const bf16_t* __restrict__ W2p, const float* __restrict__ b2,
    float* __restrict__ hout, int Nn) {
  __shared__ __align__(32) bf16_t smem[4][16 * 256];
  const int wave = threadIdx.x >> 5;
  const int lane = threadIdx.x & 31;
  const int nbase = (blockIdx.x * 4 + wave) * 16;
  if (nbase >= Nn) return;
  bf16_t* tile = smem[wave];
#pragma unroll 4
  for (int r = 0; r < 16; ++r) {
    const int n = nbase + r;
    const float4 hv = *(const float4*)(h    + (size_t)n * HDIM + lane * 4);
    const float4 av = *(const float4*)(aggN + (size_t)n * HDIM + lane * 4);
    bf16_t* dp = tile + r * 256 + lane * 4;
    dp[0] = (bf16_t)hv.x; dp[1] = (bf16_t)hv.y;
    dp[2] = (bf16_t)hv.z; dp[3] = (bf16_t)hv.w;
    dp[128] = (bf16_t)av.x; dp[129] = (bf16_t)av.y;
    dp[130] = (bf16_t)av.z; dp[131] = (bf16_t)av.w;
  }
  v8f out[8];
  mlp2_tile<8>(tile, 256, W1p, b1, W2p, b2, out);
#pragma unroll
  for (int t = 0; t < 8; ++t) {
    const int col = t * 16 + (lane & 15);
#pragma unroll
    for (int e = 0; e < 8; ++e) {
      const int row = (lane < 16) ? e : (8 + e);
      hout[(size_t)(nbase + row) * HDIM + col] = out[t][e];
    }
  }
}

__global__ void count_kernel(const int* __restrict__ batch, float* __restrict__ counts, int n) {
  int i = blockIdx.x * 256 + threadIdx.x;
  if (i < n)
    __hip_atomic_fetch_add(&counts[batch[i]], 1.0f,
                           __ATOMIC_RELAXED, __HIP_MEMORY_SCOPE_AGENT);
}

__global__ __launch_bounds__(256) void pool_kernel(const float* __restrict__ h,
                                                   const int* __restrict__ batch,
                                                   float* __restrict__ pooled, int n) {
  __shared__ float lp[NGRAPH * HDIM];
  for (int i = threadIdx.x; i < NGRAPH * HDIM; i += 256) lp[i] = 0.f;
  __syncthreads();
  const size_t total = (size_t)n * HDIM;
  for (size_t i = (size_t)blockIdx.x * 256 + threadIdx.x; i < total;
       i += (size_t)gridDim.x * 256) {
    const int node = (int)(i >> 7);
    const int c    = (int)(i & 127);
    atomicAdd(&lp[batch[node] * HDIM + c], h[i]);
  }
  __syncthreads();
  for (int i = threadIdx.x; i < NGRAPH * HDIM; i += 256) {
    float v = lp[i];
    if (v != 0.f)
      __hip_atomic_fetch_add(&pooled[i], v, __ATOMIC_RELAXED, __HIP_MEMORY_SCOPE_AGENT);
  }
}

__global__ __launch_bounds__(128) void final_kernel(
    const float* __restrict__ pooled, const float* __restrict__ counts,
    const float* __restrict__ Woa, const float* __restrict__ boa,
    const float* __restrict__ Wob, const float* __restrict__ bob,
    float* __restrict__ out) {
  __shared__ float pr[NGRAPH * HDIM];
  const int tid = threadIdx.x;
  for (int i = tid; i < NGRAPH * HDIM; i += 128) {
    const int g = i >> 7;
    float cnt = counts[g]; cnt = cnt > 1.f ? cnt : 1.f;
    float v = pooled[i] / cnt;
    pr[i] = v > 0.f ? v : 0.f;
  }
  __syncthreads();
  if (tid < NGRAPH) {
    float o = 0.f;
    for (int c2 = 0; c2 < HDIM; ++c2) {
      float a = boa[c2];
      for (int c = 0; c < HDIM; ++c) a += pr[tid * HDIM + c] * Woa[c * HDIM + c2];
      a = a > 0.f ? a : 0.f;
      o += a * Wob[c2];
    }
    out[tid] = o + bob[0];
  }
}

extern "C" void kernel_launch(void* const* d_in, const int* in_sizes, int n_in,
                              void* d_out, int out_size, void* d_ws, size_t ws_size,
                              hipStream_t stream) {
  (void)in_sizes; (void)n_in; (void)out_size; (void)ws_size;
  const float* x    = (const float*)d_in[0];
  const float* rbf  = (const float*)d_in[1];
  const float* cbf  = (const float*)d_in[2];
  const float* sbf  = (const float*)d_in[3];
  const int* eidx   = (const int*)d_in[4];
  const int* l_idx  = (const int*)d_in[5];
  const int* k_idx  = (const int*)d_in[6];
  const int* batch  = (const int*)d_in[7];
  const float* W_e1a = (const float*)d_in[8];
  const float* b_e1a = (const float*)d_in[9];
  const float* W_e1b = (const float*)d_in[10];
  const float* b_e1b = (const float*)d_in[11];
  const float* W_e2a = (const float*)d_in[12];
  const float* b_e2a = (const float*)d_in[13];
  const float* W_e2b = (const float*)d_in[14];
  const float* b_e2b = (const float*)d_in[15];
  const float* W_na  = (const float*)d_in[16];
  const float* b_na  = (const float*)d_in[17];
  const float* W_nb  = (const float*)d_in[18];
  const float* b_nb  = (const float*)d_in[19];
  const float* W_oa  = (const float*)d_in[20];
  const float* b_oa  = (const float*)d_in[21];
  const float* W_ob  = (const float*)d_in[22];
  const float* b_ob  = (const float*)d_in[23];

  char* ws = (char*)d_ws;
  size_t off = 0;
  auto take = [&](size_t bytes) -> char* {
    char* p = ws + off;
    off += (bytes + 255) & ~(size_t)255;
    return p;
  };
  float* aggE   = (float*)take((size_t)NEDGE * HDIM * 4);
  float* aggN   = (float*)take((size_t)NNODE * HDIM * 4);
  float* hbuf0  = (float*)take((size_t)NNODE * HDIM * 4);
  float* hbuf1  = (float*)take((size_t)NNODE * HDIM * 4);
  float* pooled = (float*)take(NGRAPH * HDIM * 4);
  float* counts = (float*)take(NGRAPH * 4);
  bf16_t *w1a[3], *w1b[3], *w2a[3], *w2b[3], *wna[3], *wnb[3];
  for (int p = 0; p < 3; ++p) {
    w1a[p] = (bf16_t*)take((size_t)5 * 4096 * 2);
    w1b[p] = (bf16_t*)take((size_t)4 * 4096 * 2);
    w2a[p] = (bf16_t*)take((size_t)8 * 4096 * 2);
    w2b[p] = (bf16_t*)take((size_t)4 * 4096 * 2);
    wna[p] = (bf16_t*)take((size_t)8 * 4096 * 2);
    wnb[p] = (bf16_t*)take((size_t)4 * 4096 * 2);
  }

  // Pre-pack all weights into WMMA B-fragment order (bf16, zero-padded K).
  for (int p = 0; p < 3; ++p) {
    pack_w_kernel<<<(5 * 4096) / 256, 256, 0, stream>>>(W_e1a + (size_t)p * 146 * HDIM, w1a[p], 146, 5);
    pack_w_kernel<<<(4 * 4096) / 256, 256, 0, stream>>>(W_e1b + (size_t)p * 128 * HDIM, w1b[p], 128, 4);
    pack_w_kernel<<<(8 * 4096) / 256, 256, 0, stream>>>(W_e2a + (size_t)p * 256 * HDIM, w2a[p], 256, 8);
    pack_w_kernel<<<(4 * 4096) / 256, 256, 0, stream>>>(W_e2b + (size_t)p * 128 * HDIM, w2b[p], 128, 4);
    pack_w_kernel<<<(8 * 4096) / 256, 256, 0, stream>>>(W_na  + (size_t)p * 256 * HDIM, wna[p], 256, 8);
    pack_w_kernel<<<(4 * 4096) / 256, 256, 0, stream>>>(W_nb  + (size_t)p * 128 * HDIM, wnb[p], 128, 4);
  }

  const int* src = eidx;
  const int* dst = eidx + NEDGE;
  const float* hcur = x;
  float* bufs[2] = {hbuf0, hbuf1};
  for (int p = 0; p < 3; ++p) {
    float* hnext = bufs[p & 1];
    hipMemsetAsync(aggE, 0, (size_t)NEDGE * HDIM * 4, stream);
    quad_kernel<<<NQUAD / 64, 128, 0, stream>>>(hcur, rbf, cbf, sbf, l_idx, k_idx,
                                                w1a[p], b_e1a + p * HDIM,
                                                w1b[p], b_e1b + p * HDIM, aggE, NQUAD);
    hipMemsetAsync(aggN, 0, (size_t)NNODE * HDIM * 4, stream);
    edge_kernel<<<NEDGE / 64, 128, 0, stream>>>(hcur, aggE, src, dst,
                                                w2a[p], b_e2a + p * HDIM,
                                                w2b[p], b_e2b + p * HDIM, aggN, NEDGE);
    node_kernel<<<(NNODE / 16 + 3) / 4, 128, 0, stream>>>(hcur, aggN,
                                                          wna[p], b_na + p * HDIM,
                                                          wnb[p], b_nb + p * HDIM,
                                                          hnext, NNODE);
    hcur = hnext;
  }
  hipMemsetAsync(pooled, 0, NGRAPH * HDIM * 4, stream);
  hipMemsetAsync(counts, 0, NGRAPH * 4, stream);
  count_kernel<<<(NNODE + 255) / 256, 256, 0, stream>>>(batch, counts, NNODE);
  pool_kernel<<<304, 256, 0, stream>>>(hcur, batch, pooled, NNODE);
  final_kernel<<<1, 128, 0, stream>>>(pooled, counts, W_oa, b_oa, W_ob, b_ob, (float*)d_out);
}
